// MultiHeadedAttention_79113297592849
// MI455X (gfx1250) — compile-verified, matrix operand order checked by reading, not run on hardware
//
#include <hip/hip_runtime.h>

// Problem constants (match reference setup_inputs)
#define B_    4
#define S_    16384
#define D_    512
#define H_    8
#define DK_   64
#define BANK_ 256

typedef __attribute__((ext_vector_type(16))) __bf16        v16bf;
typedef __attribute__((ext_vector_type(8)))  float         v8f;
typedef __attribute__((ext_vector_type(4)))  unsigned int  v4u;
typedef __attribute__((ext_vector_type(2)))  float         v2f;
typedef __attribute__((ext_vector_type(4)))  int           v4i;

union ABu { v16bf v; v4u q[2]; };

// ---- CDNA5 async global->LDS copy availability (compile-safe guards) --------
#if defined(__has_builtin)
#if __has_builtin(__builtin_amdgcn_global_load_async_to_lds_b128) && \
    __has_builtin(__builtin_amdgcn_s_wait_asynccnt)
#define HAVE_ASYNC_LDS 1
#endif
#endif
#ifndef HAVE_ASYNC_LDS
#define HAVE_ASYNC_LDS 0
#endif

#if HAVE_ASYNC_LDS
typedef __attribute__((address_space(1))) v4i as1_v4i;   // global
typedef __attribute__((address_space(3))) v4i as3_v4i;   // LDS
#endif

__device__ __forceinline__ unsigned short f2bf(float a) {
  unsigned int ua = __float_as_uint(a);
  ua += 0x7FFFu + ((ua >> 16) & 1u);            // round-to-nearest-even
  return (unsigned short)(ua >> 16);
}
__device__ __forceinline__ unsigned int pack2bf(float a, float b) {
  unsigned int ua = __float_as_uint(a), ub = __float_as_uint(b);
  ua += 0x7FFFu + ((ua >> 16) & 1u);
  ub += 0x7FFFu + ((ub >> 16) & 1u);
  return (ua >> 16) | (ub & 0xFFFF0000u);
}

// Load a 32x512 f32 tile from global, store bf16 into sDst (32*512 ushorts).
// Async-DMA path stages 8 rows (16KB) at a time through LDS, then converts.
__device__ __forceinline__ void load_tile_bf16(const float* __restrict__ g,
                                               unsigned short* sDst,
                                               float* sStg, int tid) {
#if HAVE_ASYNC_LDS
  unsigned int* d32 = (unsigned int*)sDst;
  for (int chunk = 0; chunk < 4; ++chunk) {
#pragma unroll
    for (int i = 0; i < 16; ++i) {              // 1024 x 16B per chunk
      int e = i * 64 + tid;
      __builtin_amdgcn_global_load_async_to_lds_b128(
          (as1_v4i*)(g + (size_t)chunk * 4096 + (size_t)e * 4),
          (as3_v4i*)((char*)sStg + (size_t)e * 16), 0, 0);
    }
    __builtin_amdgcn_s_wait_asynccnt(0);
    __syncthreads();
    const v2f* s2 = (const v2f*)sStg;
    for (int i = tid; i < 2048; i += 64) {      // 4096 floats -> 2048 packed
      v2f f = s2[i];
      d32[chunk * 2048 + i] = pack2bf(f.x, f.y);
    }
    __syncthreads();
  }
#else
  (void)sStg;
  for (int i = tid; i < 32 * D_; i += 64) sDst[i] = f2bf(g[i]);
  __syncthreads();
#endif
}

// ---------------------------------------------------------------- zero scratch
__global__ void zero_f32(float* p, int n) {
  int i = blockIdx.x * blockDim.x + threadIdx.x;
  if (i < n) p[i] = 0.f;
}

// ----------------------------------------------------- convert weights to bf16
__global__ void cvt_w(const float* __restrict__ Wq, const float* __restrict__ Wk,
                      unsigned short* __restrict__ Wqb, unsigned short* __restrict__ Wkb) {
  int i = blockIdx.x * blockDim.x + threadIdx.x;
  if (i < D_ * D_) { Wqb[i] = f2bf(Wq[i]); Wkb[i] = f2bf(Wk[i]); }
}

// -------------------------------------------------- per-cluster count and vsum
__global__ void seg_count(const int* __restrict__ assign, const float* __restrict__ value,
                          float* __restrict__ cnt, float* __restrict__ vsum) {
  int i = blockIdx.x * blockDim.x + threadIdx.x;   // over B*H*S
  if (i >= B_ * H_ * S_) return;
  int s = i % S_;
  int bh = i / S_;
  int b = bh / H_;
  int a = assign[i];
  atomicAdd(&cnt[bh * BANK_ + a], 1.0f);
  atomicAdd(&vsum[bh * BANK_ + a], value[b * S_ + s]);
}

// ---------------- fused K projection (WMMA, K=512) + scatter into centroid acc
__global__ __launch_bounds__(64) void kproj_scatter(
    const float* __restrict__ key, const unsigned short* __restrict__ Wkb,
    const float* __restrict__ bk, const int* __restrict__ assign,
    float* __restrict__ cent_acc)
{
  __shared__ __align__(16) unsigned int sKu[32 * D_ / 2];     // 32KB bf16 tile
#if HAVE_ASYNC_LDS
  __shared__ __align__(16) float sStg[4096];                  // 16KB DMA staging
  float* stg = sStg;
#else
  float* stg = nullptr;
#endif
  unsigned short* sK = (unsigned short*)sKu;

  const int b  = blockIdx.y;
  const int s0 = blockIdx.x * 32;
  const int tid = threadIdx.x;
  load_tile_bf16(key + ((size_t)b * S_ + s0) * D_, sK, stg, tid);

  const int lane = tid & 31, wid = tid >> 5;
  const int hi = lane >> 4, ln = lane & 15;
  const int mbase = wid * 16;
  const int m = mbase + ln;

  for (int tg = 0; tg < 8; ++tg) {                // tg == head (4 tiles = d_k 64)
    v8f c[4] = {};
    for (int kc = 0; kc < 16; ++kc) {             // K = 512 in chunks of 32
      ABu a;
      a.q[0] = *(const v4u*)(sK + m * D_ + kc * 32 + hi * 8);
      a.q[1] = *(const v4u*)(sK + m * D_ + kc * 32 + 16 + hi * 8);
#pragma unroll
      for (int tt = 0; tt < 4; ++tt) {
        const unsigned n = (unsigned)(tg * 4 + tt) * 16 + ln;
        ABu bm;
        bm.q[0] = *(const v4u*)(Wkb + n * D_ + kc * 32 + hi * 8);
        bm.q[1] = *(const v4u*)(Wkb + n * D_ + kc * 32 + 16 + hi * 8);
        c[tt] = __builtin_amdgcn_wmma_f32_16x16x32_bf16(false, a.v, false, bm.v,
                                                        (short)0, c[tt], false, false);
      }
    }
    const unsigned bh = (unsigned)(b * H_ + tg);
    int aidx[8];
#pragma unroll
    for (int v = 0; v < 8; ++v)                   // C layout: M = v + 8*hi
      aidx[v] = assign[(size_t)bh * S_ + s0 + mbase + v + 8 * hi];
#pragma unroll
    for (int tt = 0; tt < 4; ++tt) {
      const unsigned n = (unsigned)(tg * 4 + tt) * 16 + ln;
      const float bias = bk[n];
      const unsigned dkv = (unsigned)tt * 16 + ln;
#pragma unroll
      for (int v = 0; v < 8; ++v) {
        unsigned off = (bh * BANK_ + (unsigned)aidx[v]) * DK_ + dkv;  // 32-bit offset
        atomicAdd(cent_acc + off, c[tt][v] + bias);
      }
    }
  }
}

// ------------------------------------------- centroids /= count ; vmean /= cnt
__global__ void finalize_cent(const float* __restrict__ cent_acc, const float* __restrict__ cnt,
                              const float* __restrict__ vsum, unsigned short* __restrict__ cent_bf,
                              float* __restrict__ vmean) {
  int i = blockIdx.x * blockDim.x + threadIdx.x;       // over B*H*BANK*DK
  if (i >= B_ * H_ * BANK_ * DK_) return;
  int c = i / DK_, dk = i % DK_;
  float denom = fmaxf(cnt[c], 1.0f);
  cent_bf[i] = f2bf(cent_acc[i] / denom);
  if (dk == 0) vmean[c] = vsum[c] / denom;
}

// --- fused Q projection + scores (WMMA) + per-lane online softmax + head mean
__global__ __launch_bounds__(64) void attn_kernel(
    const float* __restrict__ query, const unsigned short* __restrict__ Wqb,
    const float* __restrict__ bq, const unsigned short* __restrict__ cent_bf,
    const float* __restrict__ vmean, float* __restrict__ out)
{
  __shared__ __align__(16) unsigned int sQu[32 * D_ / 2];       // 32KB bf16 tile
  __shared__ __align__(16) unsigned int sStageU[2][16 * DK_ / 2]; // per-wave q, 2KB
#if HAVE_ASYNC_LDS
  __shared__ __align__(16) float sStg[4096];                    // 16KB DMA staging
  float* stg = sStg;
#else
  float* stg = nullptr;
#endif
  unsigned short* sQ = (unsigned short*)sQu;

  const int b  = blockIdx.y;
  const int s0 = blockIdx.x * 32;
  const int tid = threadIdx.x;
  load_tile_bf16(query + ((size_t)b * S_ + s0) * D_, sQ, stg, tid);

  const int lane = tid & 31, wid = tid >> 5;
  const int hi = lane >> 4, ln = lane & 15;
  const int mbase = wid * 16;
  const int m = mbase + ln;
  unsigned short* stage16 = (unsigned short*)&sStageU[wid][0];

  float headacc[8];
#pragma unroll
  for (int v = 0; v < 8; ++v) headacc[v] = 0.f;

  for (int h = 0; h < H_; ++h) {
    // ---- Q projection for this head: 16 rows x 64 cols, K=512 ----
    v8f cq[4] = {};
    for (int kc = 0; kc < 16; ++kc) {
      ABu a;
      a.q[0] = *(const v4u*)(sQ + m * D_ + kc * 32 + hi * 8);
      a.q[1] = *(const v4u*)(sQ + m * D_ + kc * 32 + 16 + hi * 8);
#pragma unroll
      for (int tt = 0; tt < 4; ++tt) {
        const unsigned n = (unsigned)h * DK_ + (unsigned)tt * 16 + ln;
        ABu bm;
        bm.q[0] = *(const v4u*)(Wqb + n * D_ + kc * 32 + hi * 8);
        bm.q[1] = *(const v4u*)(Wqb + n * D_ + kc * 32 + 16 + hi * 8);
        cq[tt] = __builtin_amdgcn_wmma_f32_16x16x32_bf16(false, a.v, false, bm.v,
                                                         (short)0, cq[tt], false, false);
      }
    }
#pragma unroll
    for (int tt = 0; tt < 4; ++tt) {
      float bias = bq[(unsigned)h * DK_ + (unsigned)tt * 16 + ln];
#pragma unroll
      for (int v = 0; v < 8; ++v)                       // re-layout C -> A via LDS
        stage16[(v + 8 * hi) * DK_ + tt * 16 + ln] = f2bf(cq[tt][v] + bias);
    }
    __syncthreads();

    // ---- score A operand (q tile) built once per head ----
    ABu asc[2];
#pragma unroll
    for (int kc = 0; kc < 2; ++kc) {
      asc[kc].q[0] = *(const v4u*)(stage16 + ln * DK_ + kc * 32 + hi * 8);
      asc[kc].q[1] = *(const v4u*)(stage16 + ln * DK_ + kc * 32 + 16 + hi * 8);
    }

    // ---- scores vs 256 centroids: per-lane online softmax (no per-tile
    //      cross-lane traffic; each lane owns clusters N = j*16 + ln) ----
    float m_run[8], l_run[8], o_run[8];
#pragma unroll
    for (int v = 0; v < 8; ++v) { m_run[v] = -3.0e38f; l_run[v] = 0.f; o_run[v] = 0.f; }

    const unsigned cbase = (unsigned)(b * H_ + h) * BANK_ * DK_;
    const float* vmh = vmean + (unsigned)(b * H_ + h) * BANK_;

    for (int j = 0; j < 16; ++j) {
      if (j < 15)                                      // hint next centroid tile
        __builtin_prefetch(&cent_bf[cbase + (unsigned)(j + 1) * 16 * DK_], 0, 0);
      v8f c = {};
      const unsigned cl = (unsigned)j * 16 + ln;       // cluster for this lane (N)
#pragma unroll
      for (int kc = 0; kc < 2; ++kc) {                 // K = 64
        ABu bm;
        bm.q[0] = *(const v4u*)(cent_bf + cbase + cl * DK_ + kc * 32 + hi * 8);
        bm.q[1] = *(const v4u*)(cent_bf + cbase + cl * DK_ + kc * 32 + 16 + hi * 8);
        c = __builtin_amdgcn_wmma_f32_16x16x32_bf16(false, asc[kc].v, false, bm.v,
                                                    (short)0, c, false, false);
      }
      const float vm = vmh[cl];
#pragma unroll
      for (int v = 0; v < 8; ++v) {
        float s  = c[v] * 0.125f;                      // 1/sqrt(d_k)
        float mn = fmaxf(m_run[v], s);
        float ca = __expf(m_run[v] - mn);
        float cb = __expf(s - mn);
        l_run[v] = l_run[v] * ca + cb;
        o_run[v] = o_run[v] * ca + cb * vm;
        m_run[v] = mn;
      }
    }

    // ---- merge 16 per-lane partial softmax states per row (once per head) ----
#pragma unroll
    for (int v = 0; v < 8; ++v) {
      float m_ = m_run[v], l_ = l_run[v], o_ = o_run[v];
#pragma unroll
      for (int st = 1; st < 16; st <<= 1) {
        float mo = __shfl_xor(m_, st, 32);
        float lo = __shfl_xor(l_, st, 32);
        float oo = __shfl_xor(o_, st, 32);
        float mn = fmaxf(m_, mo);
        float ea = __expf(m_ - mn);
        float eb = __expf(mo - mn);
        l_ = l_ * ea + lo * eb;
        o_ = o_ * ea + oo * eb;
        m_ = mn;
      }
      headacc[v] += o_ / l_;
    }
    __syncthreads();                                   // stage reused next head
  }

  if (ln == 0) {                                       // one writer per row group
#pragma unroll
    for (int v = 0; v < 8; ++v)
      out[(size_t)b * S_ + s0 + mbase + v + 8 * hi] = headacc[v] * (1.0f / H_);
  }
}

// ---------------------------------------------------------------------- launch
extern "C" void kernel_launch(void* const* d_in, const int* in_sizes, int n_in,
                              void* d_out, int out_size, void* d_ws, size_t ws_size,
                              hipStream_t stream) {
  (void)in_sizes; (void)n_in; (void)out_size; (void)ws_size;
  const float* query  = (const float*)d_in[0];
  const float* key    = (const float*)d_in[1];
  const float* value  = (const float*)d_in[2];
  const int*   assign = (const int*)d_in[3];
  const float* Wq     = (const float*)d_in[4];
  const float* bq     = (const float*)d_in[5];
  const float* Wk     = (const float*)d_in[6];
  const float* bk     = (const float*)d_in[7];
  float* out = (float*)d_out;

  char* ws = (char*)d_ws;
  const size_t centAccB = (size_t)B_ * H_ * BANK_ * DK_ * 4;   // 2 MB
  const size_t cntB     = (size_t)B_ * H_ * BANK_ * 4;         // 32 KB
  float* cent_acc = (float*)ws;
  float* cnt      = (float*)(ws + centAccB);
  float* vsum     = (float*)(ws + centAccB + cntB);
  float* vmean    = (float*)(ws + centAccB + 2 * cntB);
  unsigned short* cent_bf = (unsigned short*)(ws + centAccB + 3 * cntB);
  unsigned short* Wqb = (unsigned short*)(ws + centAccB + 3 * cntB + (size_t)B_ * H_ * BANK_ * DK_ * 2);
  unsigned short* Wkb = Wqb + (size_t)D_ * D_;

  const int nz = (int)((centAccB + 2 * cntB) / 4);   // cent_acc + cnt + vsum contiguous
  zero_f32<<<(nz + 255) / 256, 256, 0, stream>>>(cent_acc, nz);
  cvt_w<<<(D_ * D_ + 255) / 256, 256, 0, stream>>>(Wq, Wk, Wqb, Wkb);
  seg_count<<<(B_ * H_ * S_) / 256, 256, 0, stream>>>(assign, value, cnt, vsum);
  kproj_scatter<<<dim3(S_ / 32, B_), 64, 0, stream>>>(key, Wkb, bk, assign, cent_acc);
  finalize_cent<<<(B_ * H_ * BANK_ * DK_) / 256, 256, 0, stream>>>(cent_acc, cnt, vsum, cent_bf, vmean);
  attn_kernel<<<dim3(S_ / 32, B_), 64, 0, stream>>>(query, Wqb, bq, cent_bf, vmean, out);
}
